// PsiQRHAttention_773094113768
// MI455X (gfx1250) — compile-verified, matrix operand order checked by reading, not run on hardware
//
#include <hip/hip_runtime.h>
#include <hip/hip_bf16.h>
#include <math.h>

// ---------------------------------------------------------------------------
// PsiQRH spectral quaternion attention for MI455X (gfx1250, wave32, WMMA).
//
// Structure:
//   1) quat_expand_kernel : W[4,out_q,in_q] f32 -> dense bf16 W^T [N=4*out_q, K=4*in_q]
//   2) quat_gemm_kernel   : C[M,N] f32 = A[M,K] f32 * W^T(bf16) via
//                           v_wmma_f32_16x16x32_bf16, 128x128x32 tiles, 8 waves,
//                           double-buffered LDS + global_prefetch_b8 pipelining
//   3) spectral_attn_kernel: per-channel 2048-pt Stockham FFT in LDS
//                           (z = Q + iK pack trick), filter, IFFT, * V
// ---------------------------------------------------------------------------

typedef __bf16 bf16_t;
typedef __bf16 v16bf __attribute__((ext_vector_type(16)));
typedef __bf16 v8bf  __attribute__((ext_vector_type(8)));
typedef float  v8f   __attribute__((ext_vector_type(8)));
typedef float  v4f   __attribute__((ext_vector_type(4)));

#define BQN    4            // batch
#define SEQ    2048         // sequence length
#define DMODEL 512
#define D4V    2048         // 4*d_model
#define MROWS  (BQN*SEQ)    // 8192 GEMM rows

// ========================= 1) quaternion weight expansion ===================
// out[n] = sum_k x[k] * Wexp[k][n]; we store Wt[n][k] (row-major N x K, bf16)
__global__ __launch_bounds__(256)
void quat_expand_kernel(const float* __restrict__ W, unsigned short* __restrict__ Wb_,
                        int out_q, int in_q) {
  bf16_t* Wb = reinterpret_cast<bf16_t*>(Wb_);
  const int Kc = 4 * in_q;
  const long long tot = (long long)(4 * out_q) * Kc;
  long long idx = (long long)blockIdx.x * blockDim.x + threadIdx.x;
  if (idx >= tot) return;
  int n = (int)(idx / Kc);
  int k = (int)(idx - (long long)n * Kc);
  int o = n >> 2, co = n & 3;
  int i = k >> 2, ci = k & 3;
  // Hamilton product tables: rows = output component, cols = input component
  const int   comp[4][4] = {{0,1,2,3},{1,0,3,2},{2,3,0,1},{3,2,1,0}};
  const float sg[4][4]   = {{ 1.f,-1.f,-1.f,-1.f},
                            { 1.f, 1.f, 1.f,-1.f},
                            { 1.f,-1.f, 1.f, 1.f},
                            { 1.f, 1.f,-1.f, 1.f}};
  float w = sg[co][ci] * W[((size_t)comp[co][ci] * out_q + o) * in_q + i];
  Wb[idx] = (bf16_t)w;
}

// ========================= 2) WMMA GEMM =====================================
// C[M,N] (f32) = A[M,K] (f32, converted to bf16 on stage-in) x B (bf16, given
// as Bt[N,K] row-major) + bias[N].  Block tile 128x128, K-step 32.
// Software-pipelined: global loads of chunk i+1 issued before LDS fragment
// reads of chunk i; LDS double-buffered so only ONE barrier per k-step.
#define BM   128
#define BN   128
#define BKK  32
#define LDSW 40   // padded row stride in halves (80B: 16B-aligned, conflict-free)

__global__ __launch_bounds__(256)
void quat_gemm_kernel(const float* __restrict__ A,
                      const unsigned short* __restrict__ Bt_,
                      const float* __restrict__ bias,
                      float* __restrict__ C,
                      int Mdim, int Ndim, int Kdim) {
  const bf16_t* Bt = reinterpret_cast<const bf16_t*>(Bt_);
  __shared__ bf16_t As[2][BM * LDSW];
  __shared__ bf16_t Bs[2][BN * LDSW];

  const int t    = threadIdx.x;
  const int lane = t & 31;
  const int wave = t >> 5;           // 8 waves
  const int wm   = (wave & 3) * 32;  // wave M offset within block tile
  const int wn   = (wave >> 2) * 64; // wave N offset within block tile
  const int lr   = lane & 15;        // row/col within 16x16 tile
  const int kh   = (lane >> 4) * 8;  // K half select per WMMA bf16 layout

  const int bm0 = blockIdx.x * BM;
  const int bn0 = blockIdx.y * BN;

  // staging: 256 threads x 16 elems = 128x32 tile
  const int srow = t >> 1;
  const int scol = (t & 1) * 16;
  const float*  arow = A  + (size_t)(bm0 + srow) * Kdim + scol;
  const bf16_t* brow = Bt + (size_t)(bn0 + srow) * Kdim + scol;

  // register staging for the in-flight chunk
  v4f  ga0, ga1, ga2, ga3;
  v8bf gb0, gb1;

  auto gload = [&](int kb) {
    const float* ap = arow + kb;
    ga0 = *(const v4f*)(ap + 0);
    ga1 = *(const v4f*)(ap + 4);
    ga2 = *(const v4f*)(ap + 8);
    ga3 = *(const v4f*)(ap + 12);
    const bf16_t* bp = brow + kb;
    gb0 = *(const v8bf*)(bp);
    gb1 = *(const v8bf*)(bp + 8);
  };
  auto lstore = [&](int bi) {
    v8bf lo, hi;
#pragma unroll
    for (int i = 0; i < 4; ++i) {
      lo[i]     = (bf16_t)ga0[i];
      lo[4 + i] = (bf16_t)ga1[i];
      hi[i]     = (bf16_t)ga2[i];
      hi[4 + i] = (bf16_t)ga3[i];
    }
    *(v8bf*)&As[bi][srow * LDSW + scol]     = lo;
    *(v8bf*)&As[bi][srow * LDSW + scol + 8] = hi;
    *(v8bf*)&Bs[bi][srow * LDSW + scol]     = gb0;
    *(v8bf*)&Bs[bi][srow * LDSW + scol + 8] = gb1;
  };

  v8f acc[2][4] = {};

  const int nk = Kdim / BKK;
  gload(0);
  lstore(0);
  __syncthreads();

  for (int i = 0; i < nk; ++i) {
    const int cur = i & 1;

    // issue next chunk's global loads early (latency hidden behind WMMAs)
    if (i + 1 < nk) gload((i + 1) * BKK);
    // warm L2/WGP$ one more chunk ahead (gfx1250 global_prefetch_b8)
    if (i + 2 < nk) {
      __builtin_prefetch(arow + (i + 2) * BKK, 0, 3);
      __builtin_prefetch(brow + (i + 2) * BKK, 0, 3);
    }

    // ---- load fragments (CDNA5 16-bit A layout: lanes 0-15 K[0:8)+K[16:24),
    //      lanes 16-31 K[8:16)+K[24:32)) ----
    v16bf afr[2], bfr[4];
#pragma unroll
    for (int tm = 0; tm < 2; ++tm) {
      const bf16_t* p = &As[cur][(wm + tm * 16 + lr) * LDSW + kh];
      v8bf lo = *(const v8bf*)p;
      v8bf hi = *(const v8bf*)(p + 16);
#pragma unroll
      for (int q = 0; q < 8; ++q) { afr[tm][q] = lo[q]; afr[tm][8 + q] = hi[q]; }
    }
#pragma unroll
    for (int tn = 0; tn < 4; ++tn) {
      const bf16_t* p = &Bs[cur][(wn + tn * 16 + lr) * LDSW + kh];
      v8bf lo = *(const v8bf*)p;
      v8bf hi = *(const v8bf*)(p + 16);
#pragma unroll
      for (int q = 0; q < 8; ++q) { bfr[tn][q] = lo[q]; bfr[tn][8 + q] = hi[q]; }
    }

    // write next chunk into the other LDS buffer (no conflict with reads of cur)
    if (i + 1 < nk) lstore(cur ^ 1);

    // ---- 8 WMMAs per wave per k-step ----
#pragma unroll
    for (int tm = 0; tm < 2; ++tm)
#pragma unroll
      for (int tn = 0; tn < 4; ++tn)
        acc[tm][tn] = __builtin_amdgcn_wmma_f32_16x16x32_bf16(
            false, afr[tm], false, bfr[tn], (short)0, acc[tm][tn], false, false);

    __syncthreads();   // single barrier per k-step (double-buffered)
  }

  // ---- epilogue: C/D layout -> lane (M = g + 8*(lane>>4), N = lane&15) ----
#pragma unroll
  for (int tm = 0; tm < 2; ++tm) {
    const int gm0 = bm0 + wm + tm * 16 + (lane >> 4) * 8;
#pragma unroll
    for (int tn = 0; tn < 4; ++tn) {
      const int gn = bn0 + wn + tn * 16 + lr;
      const float bv = bias[gn];
#pragma unroll
      for (int g = 0; g < 8; ++g)
        C[(size_t)(gm0 + g) * Ndim + gn] = acc[tm][tn][g] + bv;
    }
  }
}

// ========================= 3) spectral attention ============================
// One workgroup per (b, h, d) channel. z = Q + i*K packed FFT, combine with
// adaptive spectral filter, inverse FFT, real part * V.
#define FFTN 2048
#define FFTT 256

__global__ __launch_bounds__(256)
void spectral_attn_kernel(const float* __restrict__ Q, const float* __restrict__ Kq,
                          const float* __restrict__ V, const float* __restrict__ alpha,
                          float* __restrict__ outAttn) {
  __shared__ float2 buf[2][FFTN];   // 32 KB ping-pong (Stockham autosort)
  const int t  = threadIdx.x;
  const int wg = blockIdx.x;            // 0 .. BQN*D4V-1
  const int ch = wg & (D4V - 1);        // h*HD + d
  const int b  = wg >> 11;              // D4V == 2048
  const size_t base = (size_t)b * SEQ * D4V + ch;
  const float aval = alpha[ch];

  for (int s = t; s < FFTN; s += FFTT) {
    const size_t g = base + (size_t)s * D4V;
    buf[0][s] = make_float2(Q[g], Kq[g]);
  }
  __syncthreads();

  int cur = 0;
  auto fft_pass = [&](float sgn) {   // sgn = -1 forward, +1 inverse
#pragma unroll 1
    for (int st = 0; st < 11; ++st) {
      const int m = 1 << st;
#pragma unroll
      for (int r = 0; r < 4; ++r) {
        const int idx = t + r * FFTT;        // 0..1023 butterflies
        const int j   = idx >> st;
        const int k   = idx & (m - 1);
        const int jm  = j << st;
        float2 c0 = buf[cur][k + jm];
        float2 c1 = buf[cur][k + jm + (FFTN >> 1)];
        const float ang = sgn * 3.14159265358979323846f * (float)jm * (1.0f / (FFTN / 2));
        const float cs = __cosf(ang);
        const float sn = __sinf(ang);
        float2 d0 = make_float2(c0.x + c1.x, c0.y + c1.y);
        const float ex = c0.x - c1.x, ey = c0.y - c1.y;
        float2 d1 = make_float2(ex * cs - ey * sn, ex * sn + ey * cs);
        const int o = k + (jm << 1);
        buf[cur ^ 1][o]     = d0;
        buf[cur ^ 1][o + m] = d1;
      }
      __syncthreads();
      cur ^= 1;
    }
  };

  fft_pass(-1.0f);   // forward FFT of z = Q + iK

  // ---- unpack Qf/Kf, corr = Qf*conj(Kf), multiply adaptive filter ----
#pragma unroll 1
  for (int r = 0; r < 8; ++r) {
    const int mf = t + r * FFTT;
    float2 Zm = buf[cur][mf];
    float2 Zr = buf[cur][(FFTN - mf) & (FFTN - 1)];
    const float bx = Zr.x, by = -Zr.y;            // conj(Z[N-m])
    const float qx = 0.5f * (Zm.x + bx), qy = 0.5f * (Zm.y + by);   // Qf
    const float ux = Zm.x - bx,          uy = Zm.y - by;
    const float kx = 0.5f * uy,          ky = -0.5f * ux;           // Kf
    const float cx = qx * kx + qy * ky;            // Qf * conj(Kf)
    const float cy = qy * kx - qx * ky;
    const int   fm = (mf <= FFTN / 2) ? mf : (FFTN - mf);
    const float fr = (float)fm * (1.0f / FFTN);    // |fftfreq|
    const float th = atanf(logf(fr + 1e-6f)) * aval;
    const float fc = __cosf(th), fs = __sinf(th);
    buf[cur ^ 1][mf] = make_float2(cx * fc - cy * fs, cx * fs + cy * fc);
  }
  __syncthreads();
  cur ^= 1;

  fft_pass(1.0f);    // inverse FFT

  for (int s = t; s < FFTN; s += FFTT) {
    const size_t g = base + (size_t)s * D4V;
    outAttn[g] = buf[cur][s].x * (1.0f / FFTN) * V[g];
  }
}

// ========================= host launch ======================================
extern "C" void kernel_launch(void* const* d_in, const int* in_sizes, int n_in,
                              void* d_out, int out_size, void* d_ws, size_t ws_size,
                              hipStream_t stream) {
  (void)in_sizes; (void)n_in; (void)out_size; (void)ws_size;
  const float* query = (const float*)d_in[0];
  const float* key   = (const float*)d_in[1];
  const float* value = (const float*)d_in[2];
  const float* Wq    = (const float*)d_in[3];  const float* bq   = (const float*)d_in[4];
  const float* Wk    = (const float*)d_in[5];  const float* bk   = (const float*)d_in[6];
  const float* Wv    = (const float*)d_in[7];  const float* bv   = (const float*)d_in[8];
  const float* alpha = (const float*)d_in[9];
  const float* Wint  = (const float*)d_in[10]; const float* bint = (const float*)d_in[11];
  const float* Wfin  = (const float*)d_in[12]; const float* bfin = (const float*)d_in[13];
  const float* Wout  = (const float*)d_in[14]; const float* bout = (const float*)d_in[15];
  float* out = (float*)d_out;

  char* ws = (char*)d_ws;
  size_t off = 0;
  auto take = [&](size_t bytes) -> char* {
    char* p = ws + off;
    off += (bytes + 255) & ~(size_t)255;
    return p;
  };

  unsigned short* WqE   = (unsigned short*)take((size_t)D4V * DMODEL * 2);
  unsigned short* WkE   = (unsigned short*)take((size_t)D4V * DMODEL * 2);
  unsigned short* WvE   = (unsigned short*)take((size_t)D4V * DMODEL * 2);
  unsigned short* WintE = (unsigned short*)take((size_t)D4V * D4V * 2);
  unsigned short* WfinE = (unsigned short*)take((size_t)D4V * D4V * 2);
  unsigned short* WoutE = (unsigned short*)take((size_t)D4V * D4V * 2);
  float* Qbuf = (float*)take((size_t)MROWS * D4V * 4);
  float* Kbuf = (float*)take((size_t)MROWS * D4V * 4);
  float* Vbuf = (float*)take((size_t)MROWS * D4V * 4);
  float* attn = Qbuf;   // safe alias: each FFT WG consumes its channel first
  float* t1   = Kbuf;   // Wint output (K no longer needed)
  float* t2   = Vbuf;   // Wfin output (V no longer needed)

  // ---- expand quaternion weights to dense bf16 W^T ----
  {
    const int totS = D4V * DMODEL;   // 2048 x 512
    const int totB = D4V * D4V;      // 2048 x 2048
    quat_expand_kernel<<<(totS + 255) / 256, 256, 0, stream>>>(Wq,   WqE,   DMODEL, DMODEL / 4);
    quat_expand_kernel<<<(totS + 255) / 256, 256, 0, stream>>>(Wk,   WkE,   DMODEL, DMODEL / 4);
    quat_expand_kernel<<<(totS + 255) / 256, 256, 0, stream>>>(Wv,   WvE,   DMODEL, DMODEL / 4);
    quat_expand_kernel<<<(totB + 255) / 256, 256, 0, stream>>>(Wint, WintE, DMODEL, D4V / 4);
    quat_expand_kernel<<<(totB + 255) / 256, 256, 0, stream>>>(Wfin, WfinE, DMODEL, D4V / 4);
    quat_expand_kernel<<<(totB + 255) / 256, 256, 0, stream>>>(Wout, WoutE, DMODEL, D4V / 4);
  }

  const dim3 gemmGrid(MROWS / BM, D4V / BN);   // 64 x 16
  // ---- Q/K/V projections ----
  quat_gemm_kernel<<<gemmGrid, 256, 0, stream>>>(query, WqE, bq, Qbuf, MROWS, D4V, DMODEL);
  quat_gemm_kernel<<<gemmGrid, 256, 0, stream>>>(key,   WkE, bk, Kbuf, MROWS, D4V, DMODEL);
  quat_gemm_kernel<<<gemmGrid, 256, 0, stream>>>(value, WvE, bv, Vbuf, MROWS, D4V, DMODEL);

  // ---- spectral attention (8192 channels) ----
  spectral_attn_kernel<<<BQN * D4V, FFTT, 0, stream>>>(Qbuf, Kbuf, Vbuf, alpha, attn);

  // ---- output projections ----
  quat_gemm_kernel<<<gemmGrid, 256, 0, stream>>>(attn, WintE, bint, t1,  MROWS, D4V, D4V);
  quat_gemm_kernel<<<gemmGrid, 256, 0, stream>>>(t1,   WfinE, bfin, t2,  MROWS, D4V, D4V);
  quat_gemm_kernel<<<gemmGrid, 256, 0, stream>>>(t2,   WoutE, bout, out, MROWS, D4V, D4V);
}
